// Refine_29257317220870
// MI455X (gfx1250) — matrix operand order repaired and checked by reading, not compiled, not run on hardware
//
#include <hip/hip_runtime.h>
#include <hip/hip_bf16.h>
#include <math.h>

// ---------------------------------------------------------------------------
// Problem constants (from the reference)
// ---------------------------------------------------------------------------
static constexpr int N_ENT = 20000;
static constexpr int N_REL = 256;
static constexpr int Hh    = 512;
static constexpr int Cc    = 64;
static constexpr int Ll    = 2;
static constexpr int Tt    = 8;
static constexpr int Ee    = 30000;
static constexpr int Bb    = 4096;
static constexpr int DEC_CHUNK = 1024;   // decoder rows per chunk

typedef __bf16 bf16_t;
typedef __attribute__((ext_vector_type(16))) __bf16 v16bf;
typedef __attribute__((ext_vector_type(8)))  __bf16 v8bf;
typedef __attribute__((ext_vector_type(8)))  float  v8f;

// ---------------------------------------------------------------------------
// WMMA GEMM:  C(MxN,f32) = A(MxK,bf16,row-major) @ Bt^T
//   Bt is ALWAYS stored (NxK) row-major ("transposed B").
// Requires K % 32 == 0 (all call sites: 512 / 1024 / 32768).
// Block = 256 threads = 8 wave32 waves arranged 4(m) x 2(n).
// Block tile 128x64; wave tile 32x32 -> 4 v_wmma per K-step per wave.
// Staging loads are UNGUARDED (workspace has a tail pad; garbage rows only
// feed never-stored C rows/cols), so the inner loop is branch-free.
// All global reads, LDS stores and LDS fragment reads are 16-byte (b128).
// ---------------------------------------------------------------------------
__global__ __launch_bounds__(256) void k_gemm_bf16(
    const bf16_t* __restrict__ A, const bf16_t* __restrict__ Bt,
    float* __restrict__ C, int M, int N, int K)
{
    __shared__ __align__(16) bf16_t As[128 * 40];  // [m][k], stride 40 (pad)
    __shared__ __align__(16) bf16_t Bs[64 * 40];   // [n][k], stride 40 (pad)

    const int t    = threadIdx.x;
    const int wave = t >> 5;
    const int lane = t & 31;
    const int wr   = wave & 3;        // m-tile of 32 rows
    const int wc   = wave >> 2;       // n-tile of 32 cols
    const int m0   = blockIdx.y * 128;
    const int n0   = blockIdx.x * 64;

    // staging coordinates: thread moves 2 v8bf of A and 1 v8bf of B per step
    const int sr = t >> 2;            // 0..63
    const int sc = (t & 3) * 8;       // 0/8/16/24

    v8f acc00 = {}, acc01 = {}, acc10 = {}, acc11 = {};

    const int fm = lane & 15;
    const int kh = lane >> 4;

    for (int k0 = 0; k0 < K; k0 += 32) {
        // ---- stage A tile (128x32): 2 unguarded global b128 per thread
        {
            v8bf v0 = *(const v8bf*)&A[(size_t)(m0 + sr) * K + (k0 + sc)];
            v8bf v1 = *(const v8bf*)&A[(size_t)(m0 + 64 + sr) * K + (k0 + sc)];
            *(v8bf*)&As[sr * 40 + sc]        = v0;
            *(v8bf*)&As[(64 + sr) * 40 + sc] = v1;
        }
        // ---- stage B tile (64 n x 32 k): 1 unguarded global b128 per thread
        {
            v8bf v = *(const v8bf*)&Bt[(size_t)(n0 + sr) * K + (k0 + sc)];
            *(v8bf*)&Bs[sr * 40 + sc] = v;
        }
        // prefetch next K-tile of A (emits global_prefetch_b8)
        if (k0 + 32 < K)
            __builtin_prefetch(&A[(size_t)(m0 + sr) * K + (k0 + 32 + sc)], 0, 1);
        __syncthreads();

        // ---- A fragments (ISA layout): lane holds M=fm, K={kh*8+0..7, 16+kh*8+0..7}
        const bf16_t* ap0 = &As[(wr * 32 + fm) * 40 + kh * 8];
        const bf16_t* ap1 = &As[(wr * 32 + 16 + fm) * 40 + kh * 8];
        v16bf a0 = __builtin_shufflevector(*(const v8bf*)ap0, *(const v8bf*)(ap0 + 16),
                                           0, 1, 2, 3, 4, 5, 6, 7,
                                           8, 9, 10, 11, 12, 13, 14, 15);
        v16bf a1 = __builtin_shufflevector(*(const v8bf*)ap1, *(const v8bf*)(ap1 + 16),
                                           0, 1, 2, 3, 4, 5, 6, 7,
                                           8, 9, 10, 11, 12, 13, 14, 15);
        // ---- B fragments: lane holds N=col, K={kh*8+0..7, 16+kh*8+0..7}
        const bf16_t* bp0 = &Bs[(wc * 32 + fm) * 40 + kh * 8];
        const bf16_t* bp1 = &Bs[(wc * 32 + 16 + fm) * 40 + kh * 8];
        v16bf b0 = __builtin_shufflevector(*(const v8bf*)bp0, *(const v8bf*)(bp0 + 16),
                                           0, 1, 2, 3, 4, 5, 6, 7,
                                           8, 9, 10, 11, 12, 13, 14, 15);
        v16bf b1 = __builtin_shufflevector(*(const v8bf*)bp1, *(const v8bf*)(bp1 + 16),
                                           0, 1, 2, 3, 4, 5, 6, 7,
                                           8, 9, 10, 11, 12, 13, 14, 15);

        acc00 = __builtin_amdgcn_wmma_f32_16x16x32_bf16(false, a0, false, b0,
                                                        (short)0, acc00, false, false);
        acc01 = __builtin_amdgcn_wmma_f32_16x16x32_bf16(false, a0, false, b1,
                                                        (short)0, acc01, false, false);
        acc10 = __builtin_amdgcn_wmma_f32_16x16x32_bf16(false, a1, false, b0,
                                                        (short)0, acc10, false, false);
        acc11 = __builtin_amdgcn_wmma_f32_16x16x32_bf16(false, a1, false, b1,
                                                        (short)0, acc11, false, false);
        __syncthreads();
    }

    // ---- store D: local M = 8*kh + j within each 16-row fragment
    const int fn   = lane & 15;
    const int rb0  = m0 + wr * 32 + kh * 8;        // fragment a0 rows
    const int rb1  = rb0 + 16;                     // fragment a1 rows
    const int cb0  = n0 + wc * 32 + fn;            // fragment b0 cols
    const int cb1  = cb0 + 16;                     // fragment b1 cols
    #pragma unroll
    for (int j = 0; j < 8; ++j) {
        int r0 = rb0 + j, r1 = rb1 + j;
        if (r0 < M) {
            if (cb0 < N) C[(size_t)r0 * N + cb0] = acc00[j];
            if (cb1 < N) C[(size_t)r0 * N + cb1] = acc01[j];
        }
        if (r1 < M) {
            if (cb0 < N) C[(size_t)r1 * N + cb0] = acc10[j];
            if (cb1 < N) C[(size_t)r1 * N + cb1] = acc11[j];
        }
    }
}

// ---------------------------------------------------------------------------
// Weight convert + transpose: in (KxN, f32) -> out (NxK, bf16).  Tiled via LDS
// so both the global read and the global write are coalesced.
// ---------------------------------------------------------------------------
__global__ __launch_bounds__(256) void k_f32_to_bf16_T(
    const float* __restrict__ in, bf16_t* __restrict__ out, int K, int N)
{
    __shared__ float tile[32][33];
    const int kb = blockIdx.y * 32;
    const int nb = blockIdx.x * 32;
    const int tx = threadIdx.x & 31;
    const int ty = threadIdx.x >> 5;          // 0..7
    #pragma unroll
    for (int i = 0; i < 32; i += 8) {
        int k = kb + ty + i, n = nb + tx;
        tile[ty + i][tx] = (k < K && n < N) ? in[(size_t)k * N + n] : 0.0f;
    }
    __syncthreads();
    #pragma unroll
    for (int i = 0; i < 32; i += 8) {
        int n = nb + ty + i, k = kb + tx;
        if (n < N && k < K) out[(size_t)n * K + k] = (bf16_t)tile[tx][ty + i];
    }
}

// ---------------------------------------------------------------------------
// Elementwise / reduction helper kernels
// ---------------------------------------------------------------------------
__global__ void k_f32_to_bf16(const float* __restrict__ in, bf16_t* __restrict__ out, size_t n) {
    size_t i = (size_t)blockIdx.x * blockDim.x + threadIdx.x;
    if (i < n) out[i] = (bf16_t)in[i];
}

// row L2-normalize in place; also writes bf16 mirror
__global__ __launch_bounds__(256) void k_row_normalize(
    float* __restrict__ x, bf16_t* __restrict__ xbf, int cols)
{
    const int row = blockIdx.x;
    float* p = x + (size_t)row * cols;
    float s = 0.0f;
    for (int c = threadIdx.x; c < cols; c += blockDim.x) { float v = p[c]; s += v * v; }
    __shared__ float red[256];
    red[threadIdx.x] = s;
    __syncthreads();
    for (int o = 128; o > 0; o >>= 1) {
        if (threadIdx.x < o) red[threadIdx.x] += red[threadIdx.x + o];
        __syncthreads();
    }
    float inv = 1.0f / (sqrtf(red[0]) + 1e-8f);
    for (int c = threadIdx.x; c < cols; c += blockDim.x) {
        float v = p[c] * inv;
        p[c] = v;
        xbf[(size_t)row * cols + c] = (bf16_t)v;
    }
}

// segment-sum scatter: acc[id[e]] += rows[gsrc[e]] (float atomics) + counts
__global__ __launch_bounds__(512) void k_gather_seg_accum(
    const float* __restrict__ rows, const int* __restrict__ gsrc,
    const int* __restrict__ segid, float* __restrict__ acc,
    float* __restrict__ cnt, int H)
{
    const int e = blockIdx.x;
    const int s = gsrc ? gsrc[e] : e;
    const int g = segid[e];
    const float* rp = rows + (size_t)s * H;
    float* ap = acc + (size_t)g * H;
    for (int h = threadIdx.x; h < H; h += blockDim.x) atomicAdd(&ap[h], rp[h]);
    if (threadIdx.x == 0) atomicAdd(&cnt[g], 1.0f);
}

// xcat[r] = [ acc[r]/max(cnt,1) , rel_embeds[r] ]  (bf16, width 2H)
__global__ __launch_bounds__(512) void k_rel_finalize(
    const float* __restrict__ acc, const float* __restrict__ cnt,
    const float* __restrict__ rel_embeds, bf16_t* __restrict__ xcat, int H)
{
    const int r = blockIdx.x;
    const float inv = 1.0f / fmaxf(cnt[r], 1.0f);
    for (int h = threadIdx.x; h < H; h += blockDim.x) {
        xcat[(size_t)r * 2 * H + h]     = (bf16_t)(acc[(size_t)r * H + h] * inv);
        xcat[(size_t)r * 2 * H + H + h] = (bf16_t)rel_embeds[(size_t)r * H + h];
    }
}

__device__ __forceinline__ float dsigmoid(float x) { return 1.0f / (1.0f + expf(-x)); }

__global__ __launch_bounds__(512) void k_gru_combine(
    const float* __restrict__ gi, const float* __restrict__ gh,
    const float* __restrict__ bih, const float* __restrict__ bhh,
    float* __restrict__ relh, bf16_t* __restrict__ relh_bf, int H)
{
    const int r = blockIdx.x;
    const float* gir = gi + (size_t)r * 3 * H;
    const float* ghr = gh + (size_t)r * 3 * H;
    for (int h = threadIdx.x; h < H; h += blockDim.x) {
        float ir = gir[h]         + bih[h];
        float iz = gir[H + h]     + bih[H + h];
        float in = gir[2 * H + h] + bih[2 * H + h];
        float hr = ghr[h]         + bhh[h];
        float hz = ghr[H + h]     + bhh[H + h];
        float hn = ghr[2 * H + h] + bhh[2 * H + h];
        float rr = dsigmoid(ir + hr);
        float zz = dsigmoid(iz + hz);
        float nn = tanhf(in + rr * hn);
        float out = (1.0f - zz) * nn + zz * relh[(size_t)r * H + h];
        relh[(size_t)r * H + h] = out;
        relh_bf[(size_t)r * H + h] = (bf16_t)out;
    }
}

// msgin[e] = bf16( h[src[e]] + relh[erel[e]] )
__global__ __launch_bounds__(512) void k_msg_input(
    const float* __restrict__ h, const float* __restrict__ relh,
    const int* __restrict__ src, const int* __restrict__ erel,
    bf16_t* __restrict__ out, int H)
{
    const int e = blockIdx.x;
    const int s = src[e], r = erel[e];
    for (int hh = threadIdx.x; hh < H; hh += blockDim.x)
        out[(size_t)e * H + hh] =
            (bf16_t)(h[(size_t)s * H + hh] + relh[(size_t)r * H + hh]);
}

// h = relu(acc/max(cnt,1) + selfout); bf16 mirror
__global__ __launch_bounds__(512) void k_ent_combine(
    const float* __restrict__ acc, const float* __restrict__ cnt,
    const float* __restrict__ selfout, float* __restrict__ h,
    bf16_t* __restrict__ h_bf, int H)
{
    const int i = blockIdx.x;
    const float inv = 1.0f / fmaxf(cnt[i], 1.0f);
    for (int hh = threadIdx.x; hh < H; hh += blockDim.x) {
        float v = acc[(size_t)i * H + hh] * inv + selfout[(size_t)i * H + hh];
        v = fmaxf(v, 0.0f);
        h[(size_t)i * H + hh] = v;
        h_bf[(size_t)i * H + hh] = (bf16_t)v;
    }
}

// u = sigmoid(gate+b); ent = u*hn + (1-u)*ent; bf16 mirror
__global__ __launch_bounds__(512) void k_timegate(
    const float* __restrict__ gate, const float* __restrict__ gate_b,
    const float* __restrict__ hn, float* __restrict__ ent,
    bf16_t* __restrict__ ent_bf, int H)
{
    const int i = blockIdx.x;
    for (int hh = threadIdx.x; hh < H; hh += blockDim.x) {
        float u = dsigmoid(gate[(size_t)i * H + hh] + gate_b[hh]);
        float v = u * hn[(size_t)i * H + hh] + (1.0f - u) * ent[(size_t)i * H + hh];
        ent[(size_t)i * H + hh] = v;
        ent_bf[(size_t)i * H + hh] = (bf16_t)v;
    }
}

// 1D 'SAME' K=3 cross-correlation over hidden dim + relu -> bf16 fc input.
// Channel i row = chX[ idxX ? idxX[gb] : gb ].  Output rows are chunk-local.
__global__ __launch_bounds__(512) void k_conv_backbone(
    const float* ch0, const int* idx0,
    const float* ch1, const int* idx1,
    const float* ch2, const int* idx2,
    int nch, const float* __restrict__ W, const float* __restrict__ bias,
    bf16_t* __restrict__ out, int b_base, int H, int C)
{
    __shared__ float rows[3 * 512];
    const int lb = blockIdx.x;
    const int gb = b_base + lb;
    const float* srcs[3] = { ch0, ch1, ch2 };
    const int*   idxs[3] = { idx0, idx1, idx2 };
    for (int i = 0; i < nch; ++i) {
        int ri = idxs[i] ? idxs[i][gb] : gb;
        const float* pr = srcs[i] + (size_t)ri * H;
        for (int x = threadIdx.x; x < H; x += blockDim.x) rows[i * H + x] = pr[x];
    }
    __syncthreads();
    for (int c = 0; c < C; ++c) {
        const float bc = bias[c];
        for (int x = threadIdx.x; x < H; x += blockDim.x) {
            float s = bc;
            #pragma unroll 3
            for (int i = 0; i < 3; ++i) {
                if (i >= nch) break;
                const float* w = W + ((size_t)c * nch + i) * 3;
                float xm = (x > 0)     ? rows[i * H + x - 1] : 0.0f;
                float xc =               rows[i * H + x];
                float xp = (x < H - 1) ? rows[i * H + x + 1] : 0.0f;
                s += w[0] * xm + w[1] * xc + w[2] * xp;
            }
            s = fmaxf(s, 0.0f);
            out[(size_t)lb * ((size_t)C * H) + (size_t)c * H + x] = (bf16_t)s;
        }
    }
}

__global__ __launch_bounds__(512) void k_bias_relu(
    const float* __restrict__ in, const float* __restrict__ bias,
    float* __restrict__ outf, bf16_t* __restrict__ outb, int cols)
{
    const int r = blockIdx.x;
    for (int c = threadIdx.x; c < cols; c += blockDim.x) {
        float v = fmaxf(in[(size_t)r * cols + c] + bias[c], 0.0f);
        outf[(size_t)r * cols + c] = v;
        outb[(size_t)r * cols + c] = (bf16_t)v;
    }
}

// ---------------------------------------------------------------------------
// Host orchestration
// ---------------------------------------------------------------------------
extern "C" void kernel_launch(void* const* d_in, const int* in_sizes, int n_in,
                              void* d_out, int out_size, void* d_ws, size_t ws_size,
                              hipStream_t stream)
{
    (void)in_sizes; (void)n_in; (void)out_size; (void)ws_size;

    const float* ent_embeds = (const float*)d_in[0];
    const float* rel_embeds = (const float*)d_in[1];
    const float* W_msg      = (const float*)d_in[2];
    const float* W_self     = (const float*)d_in[3];
    const float* gru_Wih    = (const float*)d_in[4];
    const float* gru_Whh    = (const float*)d_in[5];
    const float* gru_bih    = (const float*)d_in[6];
    const float* gru_bhh    = (const float*)d_in[7];
    const float* gate_W     = (const float*)d_in[8];
    const float* gate_b     = (const float*)d_in[9];
    const float* e_conv1 = (const float*)d_in[10]; const float* e_cb1 = (const float*)d_in[11];
    const float* e_fc1   = (const float*)d_in[12]; const float* e_fb1 = (const float*)d_in[13];
    const float* e_conv2 = (const float*)d_in[14]; const float* e_cb2 = (const float*)d_in[15];
    const float* e_fc2   = (const float*)d_in[16]; const float* e_fb2 = (const float*)d_in[17];
    const float* r_conv1 = (const float*)d_in[18]; const float* r_cb1 = (const float*)d_in[19];
    const float* r_fc1   = (const float*)d_in[20]; const float* r_fb1 = (const float*)d_in[21];
    const float* r_conv2 = (const float*)d_in[22]; const float* r_cb2 = (const float*)d_in[23];
    const float* r_fc2   = (const float*)d_in[24]; const float* r_fb2 = (const float*)d_in[25];
    const int* src  = (const int*)d_in[26];
    const int* dst  = (const int*)d_in[27];
    const int* erel = (const int*)d_in[28];
    const int* subj = (const int*)d_in[29];
    const int* rel  = (const int*)d_in[30];
    const int* obj  = (const int*)d_in[31];
    float* out = (float*)d_out;

    // --- workspace carve ---
    size_t off = 0;
    auto alloc = [&](size_t bytes) -> void* {
        void* p = (char*)d_ws + off;
        off += (bytes + 255) & ~(size_t)255;
        return p;
    };
    float*  ent     = (float*) alloc((size_t)N_ENT * Hh * 4);
    bf16_t* ent_bf  = (bf16_t*)alloc((size_t)N_ENT * Hh * 2);
    float*  hbuf    = (float*) alloc((size_t)N_ENT * Hh * 4);
    bf16_t* h_bf    = (bf16_t*)alloc((size_t)N_ENT * Hh * 2);
    float*  relh    = (float*) alloc((size_t)N_REL * Hh * 4);
    bf16_t* relh_bf = (bf16_t*)alloc((size_t)N_REL * Hh * 2);
    float*  relacc  = (float*) alloc((size_t)N_REL * Hh * 4);
    float*  relcnt  = (float*) alloc((size_t)N_REL * 4);
    bf16_t* xcat    = (bf16_t*)alloc((size_t)N_REL * 2 * Hh * 2);
    float*  gi      = (float*) alloc((size_t)N_REL * 3 * Hh * 4);
    float*  gh      = (float*) alloc((size_t)N_REL * 3 * Hh * 4);
    bf16_t* msgin   = (bf16_t*)alloc((size_t)Ee * Hh * 2);
    float*  msgout  = (float*) alloc((size_t)Ee * Hh * 4);
    float*  entacc  = (float*) alloc((size_t)N_ENT * Hh * 4);
    float*  entcnt  = (float*) alloc((size_t)N_ENT * 4);
    float*  tmpNH   = (float*) alloc((size_t)N_ENT * Hh * 4);
    // transposed bf16 weights, all stored (N x K)
    bf16_t* WihT    = (bf16_t*)alloc((size_t)3 * Hh * 2 * Hh * 2);  // (1536 x 1024)
    bf16_t* WhhT    = (bf16_t*)alloc((size_t)3 * Hh * Hh * 2);      // (1536 x 512)
    bf16_t* WmsgT   = (bf16_t*)alloc((size_t)Ll * Hh * Hh * 2);     // 2 x (512 x 512)
    bf16_t* WselfT  = (bf16_t*)alloc((size_t)Ll * Hh * Hh * 2);
    bf16_t* gateWT  = (bf16_t*)alloc((size_t)Hh * Hh * 2);
    bf16_t* fcwT[4];
    for (int i = 0; i < 4; ++i) fcwT[i] = (bf16_t*)alloc((size_t)Hh * (Cc * Hh) * 2); // (512 x 32768)
    bf16_t* fcin    = (bf16_t*)alloc((size_t)DEC_CHUNK * Cc * Hh * 2);
    float*  fcout   = (float*) alloc((size_t)DEC_CHUNK * Hh * 4);
    float*  decf[4]; bf16_t* decb[4];
    for (int i = 0; i < 4; ++i) {
        decf[i] = (float*) alloc((size_t)Bb * Hh * 4);
        decb[i] = (bf16_t*)alloc((size_t)Bb * Hh * 2);
    }
    // tail pad: GEMM staging reads are unguarded and may overshoot the last
    // operand by <128 rows x K<=32768 bf16; keep that inside the arena.
    (void)alloc((size_t)8 << 20);

    auto gemm = [&](const bf16_t* A, const bf16_t* Bt, float* Cm,
                    int M, int N, int K) {
        dim3 g((N + 63) / 64, (M + 127) / 128);
        k_gemm_bf16<<<g, 256, 0, stream>>>(A, Bt, Cm, M, N, K);
    };
    auto cvtT = [&](const float* s, bf16_t* d, int K, int N) {
        dim3 g((N + 31) / 32, (K + 31) / 32);
        k_f32_to_bf16_T<<<g, 256, 0, stream>>>(s, d, K, N);
    };
    auto cvt = [&](const float* s, bf16_t* d, size_t n) {
        k_f32_to_bf16<<<(unsigned)((n + 255) / 256), 256, 0, stream>>>(s, d, n);
    };

    // --- one-time (per call) weight transpose+convert to bf16 (N x K) ---
    cvtT(gru_Wih, WihT, 2 * Hh, 3 * Hh);
    cvtT(gru_Whh, WhhT, Hh, 3 * Hh);
    for (int l = 0; l < Ll; ++l) {
        cvtT(W_msg  + (size_t)l * Hh * Hh, WmsgT  + (size_t)l * Hh * Hh, Hh, Hh);
        cvtT(W_self + (size_t)l * Hh * Hh, WselfT + (size_t)l * Hh * Hh, Hh, Hh);
    }
    cvtT(gate_W, gateWT, Hh, Hh);
    cvtT(e_fc1, fcwT[0], Cc * Hh, Hh);
    cvtT(e_fc2, fcwT[1], Cc * Hh, Hh);
    cvtT(r_fc1, fcwT[2], Cc * Hh, Hh);
    cvtT(r_fc2, fcwT[3], Cc * Hh, Hh);

    // --- init carry: ent = normalize(ent_embeds), relh = rel_embeds ---
    hipMemcpyAsync(ent, ent_embeds, (size_t)N_ENT * Hh * 4, hipMemcpyDeviceToDevice, stream);
    k_row_normalize<<<N_ENT, 256, 0, stream>>>(ent, ent_bf, Hh);
    hipMemcpyAsync(relh, rel_embeds, (size_t)N_REL * Hh * 4, hipMemcpyDeviceToDevice, stream);
    cvt(relh, relh_bf, (size_t)N_REL * Hh);

    // --- temporal recurrence ---
    for (int ts = 0; ts < Tt; ++ts) {
        const int* s = src  + (size_t)ts * Ee;
        const int* d = dst  + (size_t)ts * Ee;
        const int* r = erel + (size_t)ts * Ee;

        // relation evolution: seg-mean of incident entity embeds, GRU
        hipMemsetAsync(relacc, 0, (size_t)N_REL * Hh * 4, stream);
        hipMemsetAsync(relcnt, 0, (size_t)N_REL * 4, stream);
        k_gather_seg_accum<<<Ee, 512, 0, stream>>>(ent, s, r, relacc, relcnt, Hh);
        k_rel_finalize<<<N_REL, 512, 0, stream>>>(relacc, relcnt, rel_embeds, xcat, Hh);
        gemm(xcat,    WihT, gi, N_REL, 3 * Hh, 2 * Hh);
        gemm(relh_bf, WhhT, gh, N_REL, 3 * Hh, Hh);
        k_gru_combine<<<N_REL, 512, 0, stream>>>(gi, gh, gru_bih, gru_bhh, relh, relh_bf, Hh);

        // entity evolution: L layers of message passing
        hipMemcpyAsync(hbuf, ent,    (size_t)N_ENT * Hh * 4, hipMemcpyDeviceToDevice, stream);
        hipMemcpyAsync(h_bf, ent_bf, (size_t)N_ENT * Hh * 2, hipMemcpyDeviceToDevice, stream);
        for (int l = 0; l < Ll; ++l) {
            k_msg_input<<<Ee, 512, 0, stream>>>(hbuf, relh, s, r, msgin, Hh);
            gemm(msgin, WmsgT + (size_t)l * Hh * Hh, msgout, Ee, Hh, Hh);
            hipMemsetAsync(entacc, 0, (size_t)N_ENT * Hh * 4, stream);
            hipMemsetAsync(entcnt, 0, (size_t)N_ENT * 4, stream);
            k_gather_seg_accum<<<Ee, 512, 0, stream>>>(msgout, (const int*)nullptr, d,
                                                       entacc, entcnt, Hh);
            gemm(h_bf, WselfT + (size_t)l * Hh * Hh, tmpNH, N_ENT, Hh, Hh);
            k_ent_combine<<<N_ENT, 512, 0, stream>>>(entacc, entcnt, tmpNH, hbuf, h_bf, Hh);
        }

        // normalize + time gate
        k_row_normalize<<<N_ENT, 256, 0, stream>>>(hbuf, h_bf, Hh);
        gemm(h_bf, gateWT, tmpNH, N_ENT, Hh, Hh);
        k_timegate<<<N_ENT, 512, 0, stream>>>(tmpNH, gate_b, hbuf, ent, ent_bf, Hh);
    }

    // --- decoders (Refine ConvTransE / ConvTransR), chunked over batch ---
    auto backbone = [&](const float* c0, const int* i0,
                        const float* c1, const int* i1,
                        const float* c2, const int* i2, int nch,
                        const float* convW, const float* convB,
                        const bf16_t* fcw, const float* fcb,
                        float* outf, bf16_t* outb) {
        for (int b0 = 0; b0 < Bb; b0 += DEC_CHUNK) {
            k_conv_backbone<<<DEC_CHUNK, 512, 0, stream>>>(
                c0, i0, c1, i1, c2, i2, nch, convW, convB, fcin, b0, Hh, Cc);
            gemm(fcin, fcw, fcout, DEC_CHUNK, Hh, Cc * Hh);
            k_bias_relu<<<DEC_CHUNK, 512, 0, stream>>>(
                fcout, fcb, outf + (size_t)b0 * Hh, outb + (size_t)b0 * Hh, Hh);
        }
    };
    // o1 = backbone([se, re]),  o2 = backbone([o1, se, re])
    backbone(ent, subj, relh, rel, nullptr, nullptr, 2, e_conv1, e_cb1, fcwT[0], e_fb1,
             decf[0], decb[0]);
    backbone(decf[0], nullptr, ent, subj, relh, rel, 3, e_conv2, e_cb2, fcwT[1], e_fb2,
             decf[1], decb[1]);
    // q1 = backbone([se, oe]),  q2 = backbone([q1, se, oe])
    backbone(ent, subj, ent, obj, nullptr, nullptr, 2, r_conv1, r_cb1, fcwT[2], r_fb1,
             decf[2], decb[2]);
    backbone(decf[2], nullptr, ent, subj, ent, obj, 3, r_conv2, r_cb2, fcwT[3], r_fb2,
             decf[3], decb[3]);

    // --- logits: X @ ent^T / X @ relh^T (embedding tables already (N x K)) ---
    size_t o = 0;
    gemm(decb[0], ent_bf,  out + o, Bb, N_ENT, Hh); o += (size_t)Bb * N_ENT;
    gemm(decb[2], relh_bf, out + o, Bb, N_REL, Hh); o += (size_t)Bb * N_REL;
    gemm(decb[1], ent_bf,  out + o, Bb, N_ENT, Hh); o += (size_t)Bb * N_ENT;
    gemm(decb[3], relh_bf, out + o, Bb, N_REL, Hh);
}
